// SMAFormer_81801947120164
// MI455X (gfx1250) — matrix operand order repaired and checked by reading, not compiled, not run on hardware
//
#include <hip/hip_runtime.h>

// ============================================================================
// SMAFormer block for MI455X (gfx1250, wave32, WMMA + TDM).
// Compute-bound (~310 GFLOP vs ~0.25 GB traffic) => all GEMM-shaped work goes
// through v_wmma_f32_16x16x32_bf16 (f32->bf16 cast, f32 accumulate).
// Weights are pre-converted to bf16 once; weight A-tiles are then streamed
// into LDS by the Tensor Data Mover (tensor_load_to_lds, double-buffered,
// s_wait_tensorcnt) so the VALU does no staging work for the heavy convs.
// Attention is flash-style (online softmax) to avoid a 268MB score tensor.
// ============================================================================

#define DEV __device__ __forceinline__

typedef __attribute__((ext_vector_type(16))) __bf16 bf16x16;
typedef __attribute__((ext_vector_type(8)))  float  f32x8;
typedef unsigned int u32x4 __attribute__((ext_vector_type(4)));
typedef int          i32x4 __attribute__((ext_vector_type(4)));
typedef int          i32x8 __attribute__((ext_vector_type(8)));

union FragBF {
  bf16x16 v;
  uint4 q[2];
  unsigned short h[16];
};

DEV unsigned short f2bf(float f) {  // f32 -> bf16, round-to-nearest-even
  unsigned int u = __builtin_bit_cast(unsigned int, f);
  unsigned int r = u + 0x7FFFu + ((u >> 16) & 1u);
  return (unsigned short)(r >> 16);
}

static constexpr int Bn = 8, Sn = 1024, Cn = 256, Hn = 8, HIDn = 1024;
static constexpr float EPSF = 1e-5f;

// ---------------------------------------------------------------------------
// TDM: async-load a 128x32 bf16 tile from a row-major [M,K] matrix (row
// stride lda elements) into LDS. D# layout per CDNA5 ISA ch.8 (8.3-8.6).
// Issued once per wave (wave 0 of the block); completion via TENSORcnt.
// Toolchain uses the 6-arg builtin: (g0,u32x4)(g1,i32x8)(g2,i32x4)(g3,i32x4)
// (g4,i32x8)(cpol,i32).
// ---------------------------------------------------------------------------
DEV void tdm_load_tile_a(const unsigned short* g, int lda, int mBase, int kb,
                         const unsigned short* ldsDst, int M, int K)
{
  unsigned long long ga = (unsigned long long)(size_t)g +
      ((unsigned long long)(unsigned)mBase * (unsigned)lda + (unsigned)kb) * 2ull;
  unsigned lds = (unsigned)(size_t)ldsDst;   // flat LDS addr low bits = LDS offset
  u32x4 g0 = { 1u,                                   // count=1 (valid descriptor)
               lds,                                  // lds_addr [63:32]
               (unsigned)ga,                         // global_addr lo
               (unsigned)((ga >> 32) & 0x1FFFFFFull) // global_addr hi (bits 120:96)
                 | (2u << 30) };                     // type=2 ("image")
  i32x8 g1 = { (int)0x10000,                               // data_size=1 -> 2 bytes
               (int)((K & 0xFFFF) << 16),                  // tensor_dim0[15:0]
               (int)(((K >> 16) & 0xFFFF) | ((M & 0xFFFF) << 16)),  // dim0 hi | dim1 lo
               (int)(((M >> 16) & 0xFFFF) | (32 << 16)),   // dim1 hi | tile_dim0=32
               (int)128,                                   // tile_dim1=128 (tile_dim2=0)
               (int)lda,                                   // tensor_dim0_stride lo
               0, 0 };
  i32x4 g2 = { 1, 1, 0, 0 };   // tensor_dim2=1, tensor_dim3=1 (unused, tile_dim3=0)
  i32x4 g3 = { 0, 0, 0, 0 };
  i32x8 g4 = { 0, 0, 0, 0, 0, 0, 0, 0 };
  __builtin_amdgcn_tensor_load_to_lds(g0, g1, g2, g3, g4, 0);
}

// ---------------------------------------------------------------------------
// Tiled WMMA GEMM: D[M,N] = A[M,K] * B[K,N] (+bias).
// BMODE: 0 = B[k*ldb+n], 1 = B[n*ldb+k], 2 = implicit im2col of a (dilated)
//        3x3 conv over a 32x32 image: k -> (ci=k/9, tap=k%9), n -> pixel.
// BIASM: 0 none, 1 bias[n], 2 bias[m].
// AW: 1 = A is pre-converted bf16 and streamed by TDM (double-buffered).
// BW: 1 = B is pre-converted bf16 (BMODE 1 weights).
// Block tile 128(M) x 64(N) x 32(K); 8 waves, each wave owns 32x32 (2x2 frags).
// ---------------------------------------------------------------------------
template<int BMODE, int BIASM, int AW, int BW>
__global__ __launch_bounds__(256) void gemm_wmma(
    const void* __restrict__ A, size_t sA, int lda,
    const void* __restrict__ Bsrc, size_t sB, int ldb,
    float* __restrict__ D, size_t sD, int ldd,
    const float* __restrict__ bias,
    int M, int N, int K, int rate)
{
  __shared__ __align__(16) unsigned short As[2][128 * 32];
  __shared__ __align__(16) unsigned short Bs[64 * 32];

  const int tid  = threadIdx.x;
  const int z    = blockIdx.z;
  const float*          Af  = (const float*)A + (size_t)z * sA;
  const unsigned short* Ah  = (const unsigned short*)A + (size_t)z * sA;
  const float*          Bf  = (const float*)Bsrc + (size_t)z * sB;
  const unsigned short* Bh  = (const unsigned short*)Bsrc + (size_t)z * sB;
  float* Db = D + (size_t)z * sD;

  const int mBase = blockIdx.y * 128;
  const int nBase = blockIdx.x * 64;
  const int wave = tid >> 5;
  const int lane = tid & 31;
  const int half = lane >> 4;
  const int l16  = lane & 15;
  const int wm = (wave >> 1) * 32;   // 4 wave rows
  const int wn = (wave & 1) * 32;    // 2 wave cols

  f32x8 acc[2][2];
  #pragma unroll
  for (int i = 0; i < 2; ++i)
    #pragma unroll
    for (int j = 0; j < 2; ++j)
      #pragma unroll
      for (int r = 0; r < 8; ++r) acc[i][j][r] = 0.f;

  if (AW == 1 && wave == 0)
    tdm_load_tile_a(Ah, lda, mBase, 0, &As[0][0], M, K);   // prime buffer 0

  int parity = 0;
  for (int kb = 0; kb < K; kb += 32) {
    if (AW == 1) {
      if (wave == 0 && kb + 32 < K)
        tdm_load_tile_a(Ah, lda, mBase, kb + 32, &As[1 - parity][0], M, K);
      // in-order TDM completion: <=1 outstanding ==> current buffer is ready
      if (kb + 32 < K) __builtin_amdgcn_s_wait_tensorcnt(1);
      else             __builtin_amdgcn_s_wait_tensorcnt(0);
    } else {
      // ---- stage A tile [128][32], f32 -> bf16 ----
      #pragma unroll
      for (int e = 0; e < 16; ++e) {
        int idx = tid + 256 * e;
        int ml = idx >> 5, kl = idx & 31;
        int m = mBase + ml, k = kb + kl;
        float v = (m < M && k < K) ? Af[(size_t)m * lda + k] : 0.f;
        As[0][ml * 32 + kl] = f2bf(v);
      }
    }
    // ---- stage B tile as [n][k] bf16 ----
    #pragma unroll
    for (int e = 0; e < 8; ++e) {
      int idx = tid + 256 * e;
      int nl = idx >> 5, kl = idx & 31;
      int n = nBase + nl, k = kb + kl;
      unsigned short hv = 0;
      if (n < N && k < K) {
        if (BMODE == 1 && BW == 1) {
          hv = Bh[(size_t)n * ldb + k];
        } else {
          float v = 0.f;
          if (BMODE == 0) v = Bf[(size_t)k * ldb + n];
          else if (BMODE == 1) v = Bf[(size_t)n * ldb + k];
          else {
            int ci = k / 9, t = k - ci * 9;
            int yy = (n >> 5) + (t / 3 - 1) * rate;
            int xx = (n & 31) + (t % 3 - 1) * rate;
            if (yy >= 0 && yy < 32 && xx >= 0 && xx < 32)
              v = Bf[(size_t)ci * ldb + (yy << 5) + xx];
          }
          hv = f2bf(v);
        }
      }
      Bs[nl * 32 + kl] = hv;
    }
    __syncthreads();

    // ---- fragments per ISA 7.12.2 layouts, 2x b128 LDS reads each ----
    const unsigned short* Acur = AW ? &As[parity][0] : &As[0][0];
    FragBF af[2], bf[2];
    #pragma unroll
    for (int i = 0; i < 2; ++i) {
      const uint4* p = (const uint4*)(Acur + (wm + 16 * i + l16) * 32);
      af[i].q[0] = p[half];       // K = half*8 .. +7
      af[i].q[1] = p[2 + half];   // K = 16+half*8 .. +7
    }
    #pragma unroll
    for (int j = 0; j < 2; ++j) {
      const uint4* p = (const uint4*)(Bs + (wn + 16 * j + l16) * 32);
      bf[j].q[0] = p[half];       // K = half*8 .. +7
      bf[j].q[1] = p[2 + half];   // K = 16+half*8 .. +7
    }
    #pragma unroll
    for (int i = 0; i < 2; ++i)
      #pragma unroll
      for (int j = 0; j < 2; ++j)
        acc[i][j] = __builtin_amdgcn_wmma_f32_16x16x32_bf16(
            false, af[i].v, false, bf[j].v, (short)0, acc[i][j], false, false);
    __syncthreads();
    parity ^= 1;
  }

  // ---- epilogue (C/D layout: vgpr r -> row r+8*half, col = lane&15) ----
  #pragma unroll
  for (int i = 0; i < 2; ++i)
    #pragma unroll
    for (int j = 0; j < 2; ++j) {
      int colg = nBase + wn + 16 * j + l16;
      if (colg >= N) continue;
      #pragma unroll
      for (int r = 0; r < 8; ++r) {
        int rowg = mBase + wm + 16 * i + r + 8 * half;
        if (rowg >= M) continue;
        float v = acc[i][j][r];
        if (BIASM == 1) v += bias[colg];
        else if (BIASM == 2) v += bias[rowg];
        Db[(size_t)rowg * ldd + colg] = v;
      }
    }
}

// ---------------------------------------------------------------------------
// Flash attention: per wave: 16 query rows, loop keys in 32-tiles.
// qkv layout [B,S,3C] (q|k|v per head at col offsets 0/256/512 + h*32).
// ctx out [B,S,C].
// ---------------------------------------------------------------------------
__global__ __launch_bounds__(256) void flash_attn(const float* __restrict__ qkv,
                                                  float* __restrict__ ctx)
{
  __shared__ __align__(16) float          Sbuf[8][16 * 32];
  __shared__ __align__(16) unsigned short Pbuf[8][16 * 32];
  __shared__ float rowm[8][16], rowl[8][16], alpha[8][16];

  const int tid = threadIdx.x, wave = tid >> 5, lane = tid & 31;
  const int half = lane >> 4, l16 = lane & 15;
  const int b = blockIdx.y >> 3, h = blockIdx.y & 7;
  const int q0 = (blockIdx.x * 8 + wave) * 16;
  const float scale = 0.17677669529663687f;   // 1/sqrt(32)

  // Q fragment (A layout): lane's element e holds K = (e&7) + ((e>>3)<<4) + half*8
  FragBF qf;
  {
    size_t base = ((size_t)(b * Sn + q0 + l16)) * (3 * Cn) + h * 32;
    #pragma unroll
    for (int e = 0; e < 16; ++e) {
      int k = (e & 7) + ((e >> 3) << 4) + half * 8;
      qf.h[e] = f2bf(qkv[base + k]);
    }
  }
  if (lane < 16) { rowm[wave][lane] = -1e30f; rowl[wave][lane] = 0.f; }
  __syncthreads();

  f32x8 o0, o1;
  #pragma unroll
  for (int r = 0; r < 8; ++r) { o0[r] = 0.f; o1[r] = 0.f; }

  for (int kb = 0; kb < Sn; kb += 32) {
    // K^T fragments (B layout, contiguous d per key row)
    FragBF kf[2];
    #pragma unroll
    for (int j = 0; j < 2; ++j) {
      size_t base = ((size_t)(b * Sn + kb + 16 * j + l16)) * (3 * Cn) + Cn + h * 32 + half * 8;
      #pragma unroll
      for (int e = 0; e < 16; ++e) kf[j].h[e] = f2bf(qkv[base + (e & 7) + ((e >> 3) << 4)]);
    }
    f32x8 s0, s1;
    #pragma unroll
    for (int r = 0; r < 8; ++r) { s0[r] = 0.f; s1[r] = 0.f; }
    s0 = __builtin_amdgcn_wmma_f32_16x16x32_bf16(false, qf.v, false, kf[0].v, (short)0, s0, false, false);
    s1 = __builtin_amdgcn_wmma_f32_16x16x32_bf16(false, qf.v, false, kf[1].v, (short)0, s1, false, false);

    #pragma unroll
    for (int r = 0; r < 8; ++r) {
      int row = r + 8 * half;
      Sbuf[wave][row * 32 + l16]      = s0[r];
      Sbuf[wave][row * 32 + 16 + l16] = s1[r];
    }
    __syncthreads();

    if (lane < 16) {  // online softmax per row
      int row = lane;
      float mold = rowm[wave][row], mnew = mold;
      for (int c = 0; c < 32; ++c)
        mnew = fmaxf(mnew, Sbuf[wave][row * 32 + c] * scale);
      float al = __expf(mold - mnew), ls = 0.f;
      for (int c = 0; c < 32; ++c) {
        float p = __expf(Sbuf[wave][row * 32 + c] * scale - mnew);
        ls += p;
        Pbuf[wave][row * 32 + c] = f2bf(p);
      }
      rowl[wave][row] = rowl[wave][row] * al + ls;
      rowm[wave][row] = mnew;
      alpha[wave][row] = al;
    }
    __syncthreads();

    #pragma unroll
    for (int r = 0; r < 8; ++r) {
      float a = alpha[wave][r + 8 * half];
      o0[r] *= a; o1[r] *= a;
    }
    // P fragment (A layout) straight from LDS row-major 16x32 bf16
    FragBF pf;
    {
      const uint4* p = (const uint4*)(&Pbuf[wave][l16 * 32]);
      pf.q[0] = p[half];
      pf.q[1] = p[2 + half];
    }
    // V fragments (B layout: key index = (e&7)+((e>>3)<<4)+half*8, d col = 16j + l16)
    FragBF vf[2];
    #pragma unroll
    for (int j = 0; j < 2; ++j)
      #pragma unroll
      for (int e = 0; e < 16; ++e) {
        int k = (e & 7) + ((e >> 3) << 4) + half * 8;
        vf[j].h[e] = f2bf(qkv[((size_t)(b * Sn + kb + k)) * (3 * Cn) + 2 * Cn + h * 32 + 16 * j + l16]);
      }
    o0 = __builtin_amdgcn_wmma_f32_16x16x32_bf16(false, pf.v, false, vf[0].v, (short)0, o0, false, false);
    o1 = __builtin_amdgcn_wmma_f32_16x16x32_bf16(false, pf.v, false, vf[1].v, (short)0, o1, false, false);
    __syncthreads();
  }

  #pragma unroll
  for (int r = 0; r < 8; ++r) {
    int row = r + 8 * half;
    float inv = 1.f / rowl[wave][row];
    size_t base = ((size_t)(b * Sn + q0 + row)) * Cn + h * 32;
    ctx[base + l16]      = o0[r] * inv;
    ctx[base + 16 + l16] = o1[r] * inv;
  }
}

// ---------------------------------------------------------------------------
// LayerNorm over C of (Xa + Xb), arbitrary strides, out contiguous [B,S,C].
// ---------------------------------------------------------------------------
__global__ __launch_bounds__(256) void ln_add(
    const float* __restrict__ Xa, size_t aB, size_t aS, size_t aC,
    const float* __restrict__ Xb, size_t bB, size_t bS, size_t bC,
    const float* __restrict__ g, const float* __restrict__ be,
    float* __restrict__ out)
{
  __shared__ float red[256];
  int bs = blockIdx.x;
  int b = bs >> 10, s = bs & 1023;
  int c = threadIdx.x;
  float v = Xa[(size_t)b * aB + (size_t)s * aS + (size_t)c * aC]
          + Xb[(size_t)b * bB + (size_t)s * bS + (size_t)c * bC];
  red[c] = v; __syncthreads();
  for (int off = 128; off > 0; off >>= 1) { if (c < off) red[c] += red[c + off]; __syncthreads(); }
  float mu = red[0] * (1.f / 256.f); __syncthreads();
  float d = v - mu;
  red[c] = d * d; __syncthreads();
  for (int off = 128; off > 0; off >>= 1) { if (c < off) red[c] += red[c + off]; __syncthreads(); }
  float var = red[0] * (1.f / 256.f);
  out[(size_t)bs * 256 + c] = d * rsqrtf(var + EPSF) * g[c] + be[c];
}

// --------------------------- elementwise kernels ---------------------------
__global__ void ew_cvt_bf16(const float* __restrict__ src,
                            unsigned short* __restrict__ dst, int n)
{
  int i = blockIdx.x * 256 + threadIdx.x;
  if (i < n) dst[i] = f2bf(src[i]);
}

__global__ void ew_pa_sigmul(const float* __restrict__ x, float* __restrict__ pa,
                             const float* g, const float* bb, const float* m,
                             const float* vv, int total)
{
  int i = blockIdx.x * 256 + threadIdx.x; if (i >= total) return;
  int c = (i >> 10) & 255;
  float s = g[c] * rsqrtf(vv[c] + EPSF);
  float bn = (pa[i] - m[c]) * s + bb[c];
  pa[i] = x[i] * (1.f / (1.f + __expf(-bn)));
}

__global__ void ew_relu_bn(float* __restrict__ buf, const float* g, const float* bb,
                           const float* m, const float* vv, int total)
{
  int i = blockIdx.x * 256 + threadIdx.x; if (i >= total) return;
  int c = (i >> 10) & 1023;                       // 4C = 1024
  float v = fmaxf(buf[i], 0.f);
  buf[i] = (v - m[c]) * (g[c] * rsqrtf(vv[c] + EPSF)) + bb[c];
}

__global__ void ew_syn(float* __restrict__ outc, const float* __restrict__ x,
                       const float* g, const float* bb, const float* m,
                       const float* vv, const float* modb, int total)
{
  int i = blockIdx.x * 256 + threadIdx.x; if (i >= total) return;
  int c = (i >> 10) & 255;
  float v = (outc[i] - m[c]) * (g[c] * rsqrtf(vv[c] + EPSF)) + bb[c];
  outc[i] = v + modb[c] + x[i];
}

__global__ void ew_gelu(float* __restrict__ buf, int total)
{
  int i = blockIdx.x * 256 + threadIdx.x; if (i >= total) return;
  float v = buf[i];
  buf[i] = 0.5f * v * (1.f + erff(v * 0.70710678118654752f));
}

__global__ __launch_bounds__(256) void meanpool(const float* __restrict__ x,
                                                float* __restrict__ y0)
{
  __shared__ float red[256];
  int bc = blockIdx.x;
  float s = 0.f;
  for (int i = threadIdx.x; i < 1024; i += 256) s += x[(size_t)bc * 1024 + i];
  red[threadIdx.x] = s; __syncthreads();
  for (int off = 128; off > 0; off >>= 1) { if (threadIdx.x < off) red[threadIdx.x] += red[threadIdx.x + off]; __syncthreads(); }
  if (threadIdx.x == 0) y0[bc] = red[0] * (1.f / 1024.f);
}

__global__ void ca_fc1(const float* __restrict__ y0, const float* __restrict__ w1,
                       float* __restrict__ t16)
{
  int t = blockIdx.x * blockDim.x + threadIdx.x;   // 128 = B*16
  if (t >= Bn * 16) return;
  int b = t >> 4, j = t & 15;
  float s = 0.f;
  for (int k = 0; k < Cn; ++k) s += w1[j * Cn + k] * y0[b * Cn + k];
  t16[t] = fmaxf(s, 0.f);
}

__global__ void ca_fc2(const float* __restrict__ t16, const float* __restrict__ w2,
                       float* __restrict__ y)
{
  int i = blockIdx.x * blockDim.x + threadIdx.x;   // 2048 = B*C
  if (i >= Bn * Cn) return;
  int b = i >> 8, c = i & 255;
  float s = 0.f;
  for (int j = 0; j < 16; ++j) s += w2[c * 16 + j] * t16[b * 16 + j];
  y[i] = 1.f / (1.f + __expf(-s));
}

// ---------------------------------------------------------------------------
// Per-(b,c) modulator core: T2 = softmax((pa @ (x*y)), rows) @ sa  (all 32x32)
// ---------------------------------------------------------------------------
__global__ __launch_bounds__(256) void mod_bc(const float* __restrict__ pa,
                                              const float* __restrict__ x,
                                              const float* __restrict__ y,
                                              const float* __restrict__ saf,
                                              float* __restrict__ t2)
{
  __shared__ float Pm[1024], Xm[1024], Sm[1024], Tm[1024];
  int bc = blockIdx.x;
  size_t base = (size_t)bc * 1024;
  float yv = y[bc];
  for (int i = threadIdx.x; i < 1024; i += 256) {
    Pm[i] = pa[base + i]; Xm[i] = x[base + i]; Sm[i] = saf[base + i];
  }
  __syncthreads();
  for (int e = threadIdx.x; e < 1024; e += 256) {
    int i = e >> 5, j = e & 31;
    float s = 0.f;
    for (int k = 0; k < 32; ++k) s += Pm[i * 32 + k] * Xm[k * 32 + j];
    Tm[e] = s * yv;
  }
  __syncthreads();
  if (threadIdx.x < 32) {
    int i = threadIdx.x;
    float mx = -1e30f;
    for (int j = 0; j < 32; ++j) mx = fmaxf(mx, Tm[i * 32 + j]);
    float sum = 0.f;
    for (int j = 0; j < 32; ++j) { float p = __expf(Tm[i * 32 + j] - mx); Tm[i * 32 + j] = p; sum += p; }
    float inv = 1.f / sum;
    for (int j = 0; j < 32; ++j) Tm[i * 32 + j] *= inv;
  }
  __syncthreads();
  for (int e = threadIdx.x; e < 1024; e += 256) {
    int i = e >> 5, j = e & 31;
    float s = 0.f;
    for (int k = 0; k < 32; ++k) s += Tm[i * 32 + k] * Sm[k * 32 + j];
    t2[base + e] = s;
  }
}

// ============================================================================
// Host-side launcher (graph-capture safe: only kernel launches on stream).
// Workspace: activations (floats, with reuse) then bf16 weight pool; ~162 MB.
// ============================================================================
static constexpr size_t OFF_QKV   = 0;           // 6,291,456  [B,S,3C]
static constexpr size_t OFF_CTX   = 6291456;     // 2,097,152  [B,S,C]
static constexpr size_t OFF_X     = 8388608;     // 2,097,152  [B,C,S]
static constexpr size_t OFF_PA    = 10485760;    // 2,097,152  [B,C,S]
static constexpr size_t OFF_SACAT = 12582912;    // 8,388,608  [B,4C,S]
static constexpr size_t OFF_SAFIN = 20971520;    // 2,097,152  [B,C,S]
static constexpr size_t OFF_T2    = 23068672;    // 2,097,152  [B,C,S]
static constexpr size_t OFF_OUTC  = 25165824;    // 2,097,152  [B,C,S] (becomes syn)
static constexpr size_t OFF_Q2    = 27262976;    // 2,097,152  [B,S,C]
static constexpr size_t OFF_HT    = 0;           // 8,388,608  reuses qkv+ctx
static constexpr size_t OFF_HC1   = 12582912;    // 8,388,608  reuses sacat
static constexpr size_t OFF_HC2   = 0;           // 8,388,608  reuses hT (dead)
static constexpr size_t OFF_FFT   = 20971520;    // 2,097,152  reuses safin
static constexpr size_t OFF_Y     = 29360128;    // 2,048 [B,C]
static constexpr size_t OFF_YT    = 29362176;    // 128   [B,16]
static constexpr size_t OFF_WBF   = 29362304;    // bf16 weight pool starts here

// u16-element offsets inside the bf16 weight pool
static constexpr size_t WB_ATTN_WI = 0;          // 196,608
static constexpr size_t WB_ATTN_WO = 196608;     // 65,536
static constexpr size_t WB_MLP_W1  = 262144;     // 262,144
static constexpr size_t WB_PA_W    = 524288;     // 65,536
static constexpr size_t WB_SA_W    = 589824;     // 2,359,296
static constexpr size_t WB_SAOUT_W = 2949120;    // 262,144
static constexpr size_t WB_OUT_W   = 3211264;    // 65,536
static constexpr size_t WB_MLP_W2  = 3276800;    // 262,144
static constexpr size_t WB_DW_W    = 3538944;    // 9,437,184
static constexpr size_t WB_PW_W    = 12976128;   // 9,437,184  (end 22,413,312)

extern "C" void kernel_launch(void* const* d_in, const int* in_sizes, int n_in,
                              void* d_out, int out_size, void* d_ws, size_t ws_size,
                              hipStream_t stream) {
  const float* value    = (const float*)d_in[0];
  const float* key      = (const float*)d_in[1];
  const float* query    = (const float*)d_in[2];
  const float* res      = (const float*)d_in[3];
  const float* attn_wi  = (const float*)d_in[4];
  const float* attn_bi  = (const float*)d_in[5];
  const float* attn_wo  = (const float*)d_in[6];
  const float* attn_bo  = (const float*)d_in[7];
  const float* mod_bias = (const float*)d_in[8];
  const float* ca_w1    = (const float*)d_in[9];
  const float* ca_w2    = (const float*)d_in[10];
  const float* pa_w     = (const float*)d_in[11];
  const float* pa_bn_g  = (const float*)d_in[12];
  const float* pa_bn_b  = (const float*)d_in[13];
  const float* pa_bn_m  = (const float*)d_in[14];
  const float* pa_bn_v  = (const float*)d_in[15];
  const float* sa_w     = (const float*)d_in[16];
  const float* sa_b     = (const float*)d_in[17];
  const float* sa_bn_g  = (const float*)d_in[18];
  const float* sa_bn_b  = (const float*)d_in[19];
  const float* sa_bn_m  = (const float*)d_in[20];
  const float* sa_bn_v  = (const float*)d_in[21];
  const float* sa_out_w = (const float*)d_in[22];
  const float* sa_out_b = (const float*)d_in[23];
  const float* out_w    = (const float*)d_in[24];
  const float* out_b    = (const float*)d_in[25];
  const float* out_bn_g = (const float*)d_in[26];
  const float* out_bn_b = (const float*)d_in[27];
  const float* out_bn_m = (const float*)d_in[28];
  const float* out_bn_v = (const float*)d_in[29];
  const float* ln1_g    = (const float*)d_in[30];
  const float* ln1_b    = (const float*)d_in[31];
  const float* ln2_g    = (const float*)d_in[32];
  const float* ln2_b    = (const float*)d_in[33];
  const float* mlp_w1   = (const float*)d_in[34];
  const float* mlp_b1   = (const float*)d_in[35];
  const float* dw_w     = (const float*)d_in[36];
  const float* dw_b     = (const float*)d_in[37];
  const float* pw_w     = (const float*)d_in[38];
  const float* pw_b     = (const float*)d_in[39];
  const float* mlp_w2   = (const float*)d_in[40];
  const float* mlp_b2   = (const float*)d_in[41];

  float* ws    = (float*)d_ws;
  float* qkv   = ws + OFF_QKV;
  float* ctx   = ws + OFF_CTX;
  float* xbuf  = ws + OFF_X;
  float* pab   = ws + OFF_PA;
  float* sacat = ws + OFF_SACAT;
  float* safin = ws + OFF_SAFIN;
  float* t2    = ws + OFF_T2;
  float* outc  = ws + OFF_OUTC;
  float* q2    = ws + OFF_Q2;
  float* hT    = ws + OFF_HT;
  float* hc1   = ws + OFF_HC1;
  float* hc2   = ws + OFF_HC2;
  float* ffT   = ws + OFF_FFT;
  float* ybuf  = ws + OFF_Y;
  float* ytmp  = ws + OFF_YT;
  unsigned short* wbf = (unsigned short*)(ws + OFF_WBF);
  float* outp  = (float*)d_out;

  const dim3 blk(256);
  const size_t CS = (size_t)Cn * Sn;      // 262144
  const size_t SC = (size_t)Sn * Cn;
  const size_t HS = (size_t)HIDn * Sn;

  // --- pre-convert all weights to bf16 (once per launch; deterministic) ---
  {
    struct { const float* s; size_t off; int n; } cv[] = {
      { attn_wi,  WB_ATTN_WI, 3 * Cn * Cn },
      { attn_wo,  WB_ATTN_WO, Cn * Cn },
      { mlp_w1,   WB_MLP_W1,  HIDn * Cn },
      { pa_w,     WB_PA_W,    Cn * Cn },
      { sa_w,     WB_SA_W,    4 * Cn * Cn * 9 },
      { sa_out_w, WB_SAOUT_W, Cn * 4 * Cn },
      { out_w,    WB_OUT_W,   Cn * Cn },
      { mlp_w2,   WB_MLP_W2,  Cn * HIDn },
      { dw_w,     WB_DW_W,    HIDn * HIDn * 9 },
      { pw_w,     WB_PW_W,    HIDn * HIDn * 9 },
    };
    for (auto& c : cv)
      ew_cvt_bf16<<<(c.n + 255) / 256, blk, 0, stream>>>(c.s, wbf + c.off, c.n);
  }

  // --- QKV projections: q<-value, k<-key, v<-query (SMA arg shuffle) ---
  {
    dim3 g(Cn / 64, (Bn * Sn) / 128, 1);
    gemm_wmma<1, 1, 0, 1><<<g, blk, 0, stream>>>(value, 0, Cn, wbf + WB_ATTN_WI, 0, Cn,
                                                 qkv, 0, 3 * Cn, attn_bi,
                                                 Bn * Sn, Cn, Cn, 0);
    gemm_wmma<1, 1, 0, 1><<<g, blk, 0, stream>>>(key, 0, Cn, wbf + WB_ATTN_WI + (size_t)Cn * Cn, 0, Cn,
                                                 qkv + Cn, 0, 3 * Cn, attn_bi + Cn,
                                                 Bn * Sn, Cn, Cn, 0);
    gemm_wmma<1, 1, 0, 1><<<g, blk, 0, stream>>>(query, 0, Cn, wbf + WB_ATTN_WI + (size_t)2 * Cn * Cn, 0, Cn,
                                                 qkv + 2 * Cn, 0, 3 * Cn, attn_bi + 2 * Cn,
                                                 Bn * Sn, Cn, Cn, 0);
  }

  // --- flash attention -> ctx [B,S,C] ---
  flash_attn<<<dim3(Sn / (16 * 8), Bn * Hn), blk, 0, stream>>>(qkv, ctx);

  // --- out projection (TDM-fed weight A), transposed into CHW: x[B,C,S] ---
  gemm_wmma<1, 2, 1, 0><<<dim3(Sn / 64, Cn / 128, Bn), blk, 0, stream>>>(
      wbf + WB_ATTN_WO, 0, Cn, ctx, SC, Cn, xbuf, CS, Sn, attn_bo, Cn, Sn, Cn, 0);

  // --- Modulator: pa = x * sigmoid(bn(conv1x1(x))) ---
  gemm_wmma<0, 0, 1, 0><<<dim3(Sn / 64, Cn / 128, Bn), blk, 0, stream>>>(
      wbf + WB_PA_W, 0, Cn, xbuf, CS, Sn, pab, CS, Sn, nullptr, Cn, Sn, Cn, 0);
  ew_pa_sigmul<<<(Bn * Cn * Sn) / 256, blk, 0, stream>>>(
      xbuf, pab, pa_bn_g, pa_bn_b, pa_bn_m, pa_bn_v, Bn * Cn * Sn);

  // --- channel attention y[b,c] ---
  meanpool<<<Bn * Cn, blk, 0, stream>>>(xbuf, ybuf);
  ca_fc1<<<1, 128, 0, stream>>>(ybuf, ca_w1, ytmp);
  ca_fc2<<<(Bn * Cn) / 256, blk, 0, stream>>>(ytmp, ca_w2, ybuf);

  // --- 4 dilated 3x3 convs (implicit im2col WMMA, TDM weights) ---
  {
    const int rates[4] = {1, 6, 12, 18};
    for (int i = 0; i < 4; ++i) {
      gemm_wmma<2, 2, 1, 0><<<dim3(Sn / 64, Cn / 128, Bn), blk, 0, stream>>>(
          wbf + WB_SA_W + (size_t)i * Cn * Cn * 9, 0, Cn * 9,
          xbuf, CS, Sn,
          sacat + (size_t)i * Cn * Sn, (size_t)4 * Cn * Sn, Sn,
          sa_b + i * Cn, Cn, Sn, Cn * 9, rates[i]);
    }
  }
  ew_relu_bn<<<(Bn * 4 * Cn * Sn) / 256, blk, 0, stream>>>(
      sacat, sa_bn_g, sa_bn_b, sa_bn_m, sa_bn_v, Bn * 4 * Cn * Sn);

  // --- sa_out 1x1 conv: [4C->C] ---
  gemm_wmma<0, 2, 1, 0><<<dim3(Sn / 64, Cn / 128, Bn), blk, 0, stream>>>(
      wbf + WB_SAOUT_W, 0, 4 * Cn, sacat, (size_t)4 * Cn * Sn, Sn,
      safin, CS, Sn, sa_out_b, Cn, Sn, 4 * Cn, 0);

  // --- per-(b,c): t2 = softmax(pa @ (x*y)) @ sa ---
  mod_bc<<<Bn * Cn, blk, 0, stream>>>(pab, xbuf, ybuf, safin, t2);

  // --- out conv 1x1, BN, + mod_bias + x  (syn, in-place in outc) ---
  gemm_wmma<0, 2, 1, 0><<<dim3(Sn / 64, Cn / 128, Bn), blk, 0, stream>>>(
      wbf + WB_OUT_W, 0, Cn, t2, CS, Sn, outc, CS, Sn, out_b, Cn, Sn, Cn, 0);
  ew_syn<<<(Bn * Cn * Sn) / 256, blk, 0, stream>>>(
      outc, xbuf, out_bn_g, out_bn_b, out_bn_m, out_bn_v, mod_bias, Bn * Cn * Sn);

  // --- q2 = LN1(syn^T + res) -> [B,S,C] ---
  ln_add<<<Bn * Sn, blk, 0, stream>>>(outc, CS, 1, Sn,
                                      res, SC, Cn, 1,
                                      ln1_g, ln1_b, q2);

  // --- MLP: hT = gelu(W1 @ q2^T + b1) in [B,HID,S] ---
  gemm_wmma<1, 2, 1, 0><<<dim3(Sn / 64, HIDn / 128, Bn), blk, 0, stream>>>(
      wbf + WB_MLP_W1, 0, Cn, q2, SC, Cn, hT, HS, Sn, mlp_b1, HIDn, Sn, Cn, 0);
  ew_gelu<<<(Bn * HIDn * Sn) / 256, blk, 0, stream>>>(hT, Bn * HIDn * Sn);

  // --- two full 3x3 HIDxHID convs (the FLOP giants; TDM weight streaming) ---
  gemm_wmma<2, 2, 1, 0><<<dim3(Sn / 64, HIDn / 128, Bn), blk, 0, stream>>>(
      wbf + WB_DW_W, 0, HIDn * 9, hT, HS, Sn, hc1, HS, Sn, dw_b, HIDn, Sn, HIDn * 9, 1);
  gemm_wmma<2, 2, 1, 0><<<dim3(Sn / 64, HIDn / 128, Bn), blk, 0, stream>>>(
      wbf + WB_PW_W, 0, HIDn * 9, hc1, HS, Sn, hc2, HS, Sn, pw_b, HIDn, Sn, HIDn * 9, 1);

  // --- ffT = W2 @ hc2 + b2 -> [B,C,S] ---
  gemm_wmma<0, 2, 1, 0><<<dim3(Sn / 64, Cn / 128, Bn), blk, 0, stream>>>(
      wbf + WB_MLP_W2, 0, HIDn, hc2, HS, Sn, ffT, CS, Sn, mlp_b2, Cn, Sn, HIDn, 0);

  // --- out = LN2(ffT^T + q2) -> [B,S,C] ---
  ln_add<<<Bn * Sn, blk, 0, stream>>>(ffT, CS, 1, Sn,
                                      q2, SC, Cn, 1,
                                      ln2_g, ln2_b, outp);
}
